// Model_39084202394359
// MI455X (gfx1250) — matrix operand order repaired and checked by reading, not compiled, not run on hardware
//
#include <hip/hip_runtime.h>
#include <math.h>

#define D_MODEL   768
#define D_STATE   128
#define D_CONV    4
#define D_INNER   1536
#define NHEADS    24
#define HEADDIM   64
#define CONV_DIM  1792          // D_INNER + 2*D_STATE
#define D_IN_PROJ 3352          // 2*D_INNER + 2*D_STATE + NHEADS
#define N_LAYERS  4
#define SEQ_L     4161          // 1 + 64 + 4096
#define BATCH     4
#define NTOK      (BATCH * SEQ_L)   // 16644
#define M_PAD     16768             // NTOK rounded up to 128
#define NP_IN     3456              // D_IN_PROJ rounded up to 128
#define NP_OUT    768               // already multiple of 128
#define MEPS      1e-5f

typedef __bf16 bf16_t;
typedef bf16_t v16bf __attribute__((ext_vector_type(16)));
typedef float  v8f   __attribute__((ext_vector_type(8)));
typedef unsigned short v16us __attribute__((ext_vector_type(16)));
typedef unsigned int   v4u   __attribute__((ext_vector_type(4)));

union U32B { v16us us; v4u q[2]; };

// ---------- helpers ----------

__device__ __forceinline__ unsigned short f2bf(float f) {
  unsigned int u = __float_as_uint(f);
  u += 0x7FFFu + ((u >> 16) & 1u);      // round-to-nearest-even
  return (unsigned short)(u >> 16);
}

__device__ __forceinline__ float block_sum(float v, float* sm) {
  __syncthreads();                       // protect sm reuse across calls
  #pragma unroll
  for (int off = 16; off > 0; off >>= 1) v += __shfl_down(v, off, 32);
  const int wid = threadIdx.x >> 5;
  if ((threadIdx.x & 31) == 0) sm[wid] = v;
  __syncthreads();
  if (threadIdx.x < 32) {
    float t = (threadIdx.x < (blockDim.x >> 5)) ? sm[threadIdx.x] : 0.f;
    #pragma unroll
    for (int off = 4; off > 0; off >>= 1) t += __shfl_down(t, off, 32);
    if (threadIdx.x == 0) sm[0] = t;
  }
  __syncthreads();
  return sm[0];
}

// ---------- weight conversion: (K x N) f32 row-major -> (N_PAD x K) bf16 transposed ----------

__global__ void cvt_w_t(const float* __restrict__ src, unsigned short* __restrict__ dst,
                        int K, int N, int N_PAD) {
  long i = (long)blockIdx.x * blockDim.x + threadIdx.x;
  const long total = (long)N_PAD * K;
  const long stride = (long)gridDim.x * blockDim.x;
  for (; i < total; i += stride) {
    const int n = (int)(i / K), k = (int)(i % K);
    dst[i] = (n < N) ? f2bf(src[(long)k * N + n]) : (unsigned short)0;
  }
}

// ---------- embedding + norm0 ----------

__global__ __launch_bounds__(256)
void embed_norm0(const float* __restrict__ im8,
                 const float* __restrict__ frgb_w, const float* __restrict__ frgb_b,
                 const float* __restrict__ s0v, const float* __restrict__ suffix,
                 const float* __restrict__ n0w, const float* __restrict__ n0b,
                 float* __restrict__ x, float* __restrict__ loss_slot)
{
  __shared__ float sm[8];
  const int g = blockIdx.x;              // token id
  const int b = g / SEQ_L, t = g % SEQ_L;
  if (g == 0 && threadIdx.x == 0) *loss_slot = 0.f;

  float v[3];
  float s = 0.f;
  #pragma unroll
  for (int r = 0; r < 3; ++r) {
    const int d = threadIdx.x + r * 256;
    float val;
    if (t == 0) {
      val = s0v[d];
    } else {
      int i8;
      float extra = 0.f;
      if (t <= 64) {
        i8 = t - 1;
      } else {
        const int tp = t - 65;
        const int rr = tp >> 6, cc = tp & 63;
        i8 = ((rr >> 3) << 3) + (cc >> 3);
        extra = suffix[(long)tp * D_MODEL + d];
      }
      const float* p = &im8[(b * 64 + i8) * 3];
      val = p[0] * frgb_w[d] + p[1] * frgb_w[D_MODEL + d] +
            p[2] * frgb_w[2 * D_MODEL + d] + frgb_b[d] + extra;
    }
    v[r] = val; s += val;
  }
  const float mean = block_sum(s, sm) * (1.f / (float)D_MODEL);
  float vs = 0.f;
  #pragma unroll
  for (int r = 0; r < 3; ++r) { const float d0 = v[r] - mean; vs += d0 * d0; }
  const float var = block_sum(vs, sm) * (1.f / (float)D_MODEL);
  const float inv = rsqrtf(var + MEPS);
  #pragma unroll
  for (int r = 0; r < 3; ++r) {
    const int d = threadIdx.x + r * 256;
    x[(long)g * D_MODEL + d] = (v[r] - mean) * inv * n0w[d] + n0b[d];
  }
}

// ---------- per-layer LayerNorm -> bf16 ----------

__global__ __launch_bounds__(256)
void ln_bf16(const float* __restrict__ x,
             const float* __restrict__ w, const float* __restrict__ bb,
             unsigned short* __restrict__ xln)
{
  __shared__ float sm[8];
  const long g = blockIdx.x;
  float v[3]; float s = 0.f;
  #pragma unroll
  for (int r = 0; r < 3; ++r) {
    v[r] = x[g * D_MODEL + threadIdx.x + r * 256];
    s += v[r];
  }
  const float mean = block_sum(s, sm) * (1.f / (float)D_MODEL);
  float vs = 0.f;
  #pragma unroll
  for (int r = 0; r < 3; ++r) { const float d0 = v[r] - mean; vs += d0 * d0; }
  const float var = block_sum(vs, sm) * (1.f / (float)D_MODEL);
  const float inv = rsqrtf(var + MEPS);
  #pragma unroll
  for (int r = 0; r < 3; ++r) {
    const int d = threadIdx.x + r * 256;
    xln[g * D_MODEL + d] = f2bf((v[r] - mean) * inv * w[d] + bb[d]);
  }
}

// ---------- bf16 WMMA GEMM: C[M,N] (+)= A[M,K] @ Bt[N,K]^T ----------
// Block tile 128x128, 8 waves of 64x32. Both operands pre-padded so loads are
// unguarded and fully vectorized; single-LDS-buffer register pipelining.

template<bool ACCUM>
__global__ __launch_bounds__(256)
void gemm_bf16(const unsigned short* __restrict__ A,   // [>=blkM+128][K]
               const unsigned short* __restrict__ Bt,  // [>=blkN+128][K] (transposed weights)
               float* __restrict__ C,                  // [M][N]
               int M, int N, int K)
{
  constexpr int LS = 40;    // LDS row stride in ushorts (80B: 16B-aligned, conflict-free)
  __shared__ __align__(16) unsigned short As[128 * LS];
  __shared__ __align__(16) unsigned short Bs[128 * LS];

  const int tid  = threadIdx.x;
  const int lane = tid & 31;
  const int wid  = tid >> 5;
  const int wm   = (wid >> 2) * 64;     // wave M offset in block tile
  const int wn   = (wid & 3) * 32;      // wave N offset in block tile
  const int lr   = lane & 15;
  const int hi   = lane >> 4;           // lane half (0/1)

  const int blkM = blockIdx.y * 128;
  const int blkN = blockIdx.x * 128;

  // Fill assignment: chunk c -> row c>>2, k-offset (c&3)*8; this thread owns c=tid, tid+256.
  const int r0 = tid >> 2, kq = (tid & 3) * 8;
  const int r1 = r0 + 64;
  const unsigned short* Ag0 = A  + (long)(blkM + r0) * K + kq;
  const unsigned short* Ag1 = A  + (long)(blkM + r1) * K + kq;
  const unsigned short* Bg0 = Bt + (long)(blkN + r0) * K + kq;
  const unsigned short* Bg1 = Bt + (long)(blkN + r1) * K + kq;
  unsigned short* As0 = &As[r0 * LS + kq];
  unsigned short* As1 = &As[r1 * LS + kq];
  unsigned short* Bs0 = &Bs[r0 * LS + kq];
  unsigned short* Bs1 = &Bs[r1 * LS + kq];

  v8f acc[4][2];
  #pragma unroll
  for (int mi = 0; mi < 4; ++mi)
    #pragma unroll
    for (int ni = 0; ni < 2; ++ni)
      #pragma unroll
      for (int j = 0; j < 8; ++j) acc[mi][ni][j] = 0.f;

  // prologue loads
  v4u ra0 = *(const v4u*)Ag0;
  v4u ra1 = *(const v4u*)Ag1;
  v4u rb0 = *(const v4u*)Bg0;
  v4u rb1 = *(const v4u*)Bg1;

  for (int k0 = 0; k0 < K; k0 += 32) {
    *(v4u*)As0 = ra0; *(v4u*)As1 = ra1;
    *(v4u*)Bs0 = rb0; *(v4u*)Bs1 = rb1;
    __syncthreads();

    if (k0 + 32 < K) {          // issue next tile's global loads early (overlap with WMMA)
      ra0 = *(const v4u*)(Ag0 + k0 + 32);
      ra1 = *(const v4u*)(Ag1 + k0 + 32);
      rb0 = *(const v4u*)(Bg0 + k0 + 32);
      rb1 = *(const v4u*)(Bg1 + k0 + 32);
      if (k0 + 64 < K) {        // CDNA5 speculative prefetch two steps ahead
        __builtin_prefetch(Ag0 + k0 + 64, 0, 1);
        __builtin_prefetch(Bg0 + k0 + 64, 0, 1);
      }
    }

    // A fragments: lane = row; elements {hi*8+j, 16+hi*8+j} -> two 16B LDS loads
    v16bf afr[4];
    #pragma unroll
    for (int mi = 0; mi < 4; ++mi) {
      const unsigned short* ap = &As[(wm + mi * 16 + lr) * LS + hi * 8];
      U32B u;
      u.q[0] = *(const v4u*)ap;
      u.q[1] = *(const v4u*)(ap + 16);
      afr[mi] = __builtin_bit_cast(v16bf, u.us);
    }
    // B fragments: lane = column; elements {hi*8+j, 16+hi*8+j} -> two 16B LDS loads
    v16bf bfr[2];
    #pragma unroll
    for (int ni = 0; ni < 2; ++ni) {
      const unsigned short* bp = &Bs[(wn + ni * 16 + lr) * LS + hi * 8];
      U32B u;
      u.q[0] = *(const v4u*)bp;
      u.q[1] = *(const v4u*)(bp + 16);
      bfr[ni] = __builtin_bit_cast(v16bf, u.us);
    }

    #pragma unroll
    for (int mi = 0; mi < 4; ++mi)
      #pragma unroll
      for (int ni = 0; ni < 2; ++ni)
        acc[mi][ni] = __builtin_amdgcn_wmma_f32_16x16x32_bf16(
            false, afr[mi], false, bfr[ni], (short)0, acc[mi][ni], false, false);

    __syncthreads();
  }

  // Epilogue: C/D layout — lane lr = column, VGPR j = row j (+8 for hi lanes)
  #pragma unroll
  for (int mi = 0; mi < 4; ++mi)
    #pragma unroll
    for (int ni = 0; ni < 2; ++ni)
      #pragma unroll
      for (int j = 0; j < 8; ++j) {
        const int row = blkM + wm + mi * 16 + hi * 8 + j;
        const int col = blkN + wn + ni * 16 + lr;
        if (row < M && col < N) {
          const long idx = (long)row * N + col;
          const float v = acc[mi][ni][j];
          if (ACCUM) C[idx] += v; else C[idx] = v;
        }
      }
}

// ---------- causal depthwise conv + SiLU + dt/dA ----------

__global__ __launch_bounds__(256)
void conv_dt(const float* __restrict__ zx,
             const float* __restrict__ cw, const float* __restrict__ cb,
             const float* __restrict__ dtb, const float* __restrict__ Alog,
             float* __restrict__ xconv, float* __restrict__ dtout,
             float* __restrict__ dAout)
{
  const int g = blockIdx.x;
  const int b = g / SEQ_L, t = g % SEQ_L;
  for (int c = threadIdx.x; c < CONV_DIM; c += 256) {
    float acc = cb[c];
    #pragma unroll
    for (int k = 0; k < D_CONV; ++k) {
      const int tt = t - (D_CONV - 1) + k;
      if (tt >= 0)
        acc += cw[c * D_CONV + k] *
               zx[((long)(b * SEQ_L + tt)) * D_IN_PROJ + D_INNER + c];
    }
    acc = acc / (1.f + expf(-acc));                 // SiLU
    xconv[(long)g * CONV_DIM + c] = acc;
  }
  if (threadIdx.x < NHEADS) {
    const int h = threadIdx.x;
    const float A   = -expf(Alog[h]);
    const float raw = zx[(long)g * D_IN_PROJ + D_INNER + CONV_DIM + h] + dtb[h];
    const float dtv = (raw > 20.f) ? raw : log1pf(expf(raw));   // softplus
    dtout[(long)g * NHEADS + h] = dtv;
    dAout[(long)g * NHEADS + h] = expf(dtv * A);
  }
}

// ---------- sequential SSM scan: one block per (b, head) ----------
// state (64 x 128) f32 in registers: thread tid owns p = tid/4, n = (tid%4)*32 .. +32

__global__ __launch_bounds__(256)
void ssm_scan(const float* __restrict__ xconv,
              const float* __restrict__ dtb_, const float* __restrict__ dAb_,
              const float* __restrict__ Dvec,
              float* __restrict__ ys)
{
  __shared__ float Bs[D_STATE], Cs[D_STATE], xs[HEADDIM];
  const int b = blockIdx.x / NHEADS, h = blockIdx.x % NHEADS;
  const int tid = threadIdx.x;
  const int p  = tid >> 2;
  const int q  = tid & 3;
  const int n0 = q * 32;
  float st[32];
  #pragma unroll
  for (int j = 0; j < 32; ++j) st[j] = 0.f;
  const float Dh = Dvec[h];

  for (int t = 0; t < SEQ_L; ++t) {
    const long row = (long)(b * SEQ_L + t);
    __syncthreads();                                 // prior reads complete
    if (tid < D_STATE) Bs[tid]       = xconv[row * CONV_DIM + D_INNER + tid];
    else               Cs[tid - 128] = xconv[row * CONV_DIM + D_INNER + D_STATE + (tid - 128)];
    if (tid < HEADDIM) xs[tid]       = xconv[row * CONV_DIM + h * HEADDIM + tid];
    __syncthreads();

    const float dtt  = dtb_[row * NHEADS + h];
    const float dAt  = dAb_[row * NHEADS + h];
    const float xv   = xs[p];
    const float coef = dtt * xv;
    float acc = 0.f;
    #pragma unroll
    for (int j = 0; j < 32; ++j) {
      st[j] = st[j] * dAt + coef * Bs[n0 + j];
      acc  += st[j] * Cs[n0 + j];
    }
    // reduce over the 4 lanes (q = 0..3) that share this p
    acc += __shfl_xor(acc, 1, 32);
    acc += __shfl_xor(acc, 2, 32);
    if (q == 0)
      ys[(row * NHEADS + h) * HEADDIM + p] = acc + Dh * xv;
  }
}

// ---------- gating + RMSNorm -> bf16 ----------

__global__ __launch_bounds__(256)
void gate_rms(const float* __restrict__ ys, const float* __restrict__ zx,
              const float* __restrict__ gw, unsigned short* __restrict__ yg)
{
  __shared__ float sm[8];
  const long g = blockIdx.x;
  float vr[6]; float ss = 0.f;
  #pragma unroll
  for (int r = 0; r < 6; ++r) {
    const int idx = threadIdx.x + r * 256;
    const float y = ys[g * D_INNER + idx];
    const float z = zx[g * D_IN_PROJ + idx];        // z slice is [0, D_INNER)
    const float sv = y * (z / (1.f + expf(-z)));    // y * silu(z)
    vr[r] = sv; ss += sv * sv;
  }
  const float ms  = block_sum(ss, sm) * (1.f / (float)D_INNER);
  const float inv = rsqrtf(ms + MEPS);
  #pragma unroll
  for (int r = 0; r < 6; ++r) {
    const int idx = threadIdx.x + r * 256;
    yg[g * D_INNER + idx] = f2bf(vr[r] * inv * gw[idx]);
  }
}

// ---------- to_rgb + MSE loss ----------

__global__ __launch_bounds__(256)
void to_rgb_loss(const float* __restrict__ x, const float* __restrict__ W,
                 const float* __restrict__ bb, const float* __restrict__ im64,
                 float* __restrict__ out)
{
  __shared__ float sm[8];
  const int g = blockIdx.x;            // b*4096 + t
  const int b = g >> 12, t = g & 4095;
  const long xrow = ((long)b * SEQ_L + 65 + t) * D_MODEL;
  float p0 = 0.f, p1 = 0.f, p2 = 0.f;
  #pragma unroll
  for (int r = 0; r < 3; ++r) {
    const int d = threadIdx.x + r * 256;
    const float xv = x[xrow + d];
    p0 += xv * W[d * 3 + 0];
    p1 += xv * W[d * 3 + 1];
    p2 += xv * W[d * 3 + 2];
  }
  const float t0 = block_sum(p0, sm);
  const float t1 = block_sum(p1, sm);
  const float t2 = block_sum(p2, sm);
  if (threadIdx.x == 0) {
    const float y0 = t0 + bb[0], y1 = t1 + bb[1], y2 = t2 + bb[2];
    const long o = (long)g * 3;
    out[o] = y0; out[o + 1] = y1; out[o + 2] = y2;
    const float d0 = y0 - im64[o], d1 = y1 - im64[o + 1], d2 = y2 - im64[o + 2];
    atomicAdd(&out[(long)BATCH * 4096 * 3],
              (d0 * d0 + d1 * d1 + d2 * d2) * (1.f / (float)(BATCH * 4096 * 3)));
  }
}

// ---------- host driver ----------

extern "C" void kernel_launch(void* const* d_in, const int* in_sizes, int n_in,
                              void* d_out, int out_size, void* d_ws, size_t ws_size,
                              hipStream_t stream)
{
  (void)in_sizes; (void)n_in; (void)out_size; (void)ws_size;
  const float* im8    = (const float*)d_in[0];
  const float* im64   = (const float*)d_in[1];
  const float* frgb_w = (const float*)d_in[2];
  const float* frgb_b = (const float*)d_in[3];
  const float* trgb_w = (const float*)d_in[4];
  const float* trgb_b = (const float*)d_in[5];
  const float* s0v    = (const float*)d_in[6];
  const float* suffix = (const float*)d_in[7];
  const float* n0w    = (const float*)d_in[8];
  const float* n0b    = (const float*)d_in[9];
  const float* lnw    = (const float*)d_in[10];
  const float* lnb    = (const float*)d_in[11];
  const float* wi     = (const float*)d_in[12];
  const float* cw     = (const float*)d_in[13];
  const float* cb     = (const float*)d_in[14];
  const float* dtb    = (const float*)d_in[15];
  const float* Alog   = (const float*)d_in[16];
  const float* Dv     = (const float*)d_in[17];
  const float* gnw    = (const float*)d_in[18];
  const float* wo     = (const float*)d_in[19];

  char* p = (char*)d_ws;
  auto alloc = [&](size_t bytes) -> void* {
    void* r = (void*)p;
    p += (bytes + 255) & ~(size_t)255;
    return r;
  };
  float*          x     = (float*)alloc((size_t)NTOK * D_MODEL * 4);
  unsigned short* xln   = (unsigned short*)alloc((size_t)M_PAD * D_MODEL * 2);   // padded M
  float*          zx    = (float*)alloc((size_t)NTOK * D_IN_PROJ * 4);
  float*          xconv = (float*)alloc((size_t)NTOK * CONV_DIM * 4);
  float*          dtbuf = (float*)alloc((size_t)NTOK * NHEADS * 4);
  float*          dAbuf = (float*)alloc((size_t)NTOK * NHEADS * 4);
  float*          ysbuf = (float*)alloc((size_t)NTOK * D_INNER * 4);
  unsigned short* yg    = (unsigned short*)alloc((size_t)M_PAD * D_INNER * 2);   // padded M
  unsigned short* wib   = (unsigned short*)alloc((size_t)N_LAYERS * NP_IN  * D_MODEL * 2); // [NP_IN][K]
  unsigned short* wob   = (unsigned short*)alloc((size_t)N_LAYERS * NP_OUT * D_INNER * 2); // [NP_OUT][K]

  float* out = (float*)d_out;
  float* loss_slot = out + (size_t)BATCH * 4096 * 3;

  for (int l = 0; l < N_LAYERS; ++l) {
    cvt_w_t<<<1024, 256, 0, stream>>>(wi + (size_t)l * D_MODEL * D_IN_PROJ,
                                      wib + (size_t)l * NP_IN * D_MODEL,
                                      D_MODEL, D_IN_PROJ, NP_IN);
    cvt_w_t<<<1024, 256, 0, stream>>>(wo + (size_t)l * D_INNER * D_MODEL,
                                      wob + (size_t)l * NP_OUT * D_INNER,
                                      D_INNER, D_MODEL, NP_OUT);
  }

  embed_norm0<<<NTOK, 256, 0, stream>>>(im8, frgb_w, frgb_b, s0v, suffix,
                                        n0w, n0b, x, loss_slot);

  dim3 g1(NP_IN / 128,  M_PAD / 128);   // 27 x 131
  dim3 g2(NP_OUT / 128, M_PAD / 128);   // 6 x 131

  for (int l = 0; l < N_LAYERS; ++l) {
    ln_bf16<<<NTOK, 256, 0, stream>>>(x, lnw + l * D_MODEL, lnb + l * D_MODEL, xln);
    gemm_bf16<false><<<g1, 256, 0, stream>>>(
        xln, wib + (size_t)l * NP_IN * D_MODEL, zx, NTOK, D_IN_PROJ, D_MODEL);
    conv_dt<<<NTOK, 256, 0, stream>>>(
        zx, cw + (size_t)l * CONV_DIM * D_CONV, cb + l * CONV_DIM,
        dtb + l * NHEADS, Alog + l * NHEADS, xconv, dtbuf, dAbuf);
    ssm_scan<<<BATCH * NHEADS, 256, 0, stream>>>(xconv, dtbuf, dAbuf,
                                                 Dv + l * NHEADS, ysbuf);
    gate_rms<<<NTOK, 256, 0, stream>>>(ysbuf, zx, gnw + l * D_INNER, yg);
    gemm_bf16<true><<<g2, 256, 0, stream>>>(
        yg, wob + (size_t)l * NP_OUT * D_INNER, x, NTOK, D_MODEL, D_INNER);
  }

  to_rgb_loss<<<BATCH * 4096, 256, 0, stream>>>(x, trgb_w, trgb_b, im64, out);
}
